// DTWEncTrans_36438502539736
// MI455X (gfx1250) — matrix operand order repaired and checked by reading, not compiled, not run on hardware
//
#include <hip/hip_runtime.h>

typedef __attribute__((ext_vector_type(16))) _Float16 v16h;
typedef __attribute__((ext_vector_type(8)))  _Float16 v8h;
typedef __attribute__((ext_vector_type(8)))  float    v8f;

#define INF_F 1e30f
#define NEG_SLOPE 0.2f
#define KS  7
#define PAD 3

// concat two 8-half vectors into a 16-half WMMA fragment
__device__ __forceinline__ v16h cat8(v8h a, v8h b) {
    return __builtin_shufflevector(a, b, 0,1,2,3,4,5,6,7,8,9,10,11,12,13,14,15);
}

// ---------------------------------------------------------------------------
// Pack f32 (B,Cin,T) -> f16 (B, T+2*PAD, CinPad), reflect-padded rows,
// zero-padded channels. Row-contiguous channels == WMMA K dimension.
// ---------------------------------------------------------------------------
__global__ void pack_input(const float* __restrict__ x, _Float16* __restrict__ xT,
                           int B, int Cin, int Cpad, int T)
{
    const int Tp = T + 2 * PAD;
    int idx = blockIdx.x * blockDim.x + threadIdx.x;
    int total = B * Tp * Cpad;
    if (idx >= total) return;
    int c = idx % Cpad;
    int rem = idx / Cpad;
    int p = rem % Tp;
    int b = rem / Tp;
    float v = 0.0f;
    if (c < Cin) {
        int tin = p - PAD;
        tin = (tin < 0) ? -tin : tin;
        tin = (tin >= T) ? (2 * T - 2 - tin) : tin;
        v = x[((size_t)b * Cin + c) * T + tin];
    }
    xT[idx] = (_Float16)v;
}

// ---------------------------------------------------------------------------
// Pack weights f32 (Cout,Cin,KS) -> f16 (KS, CoutPad, CinPad), zero-padded.
// ---------------------------------------------------------------------------
__global__ void pack_weights(const float* __restrict__ W, _Float16* __restrict__ W16,
                             int Cout, int Cin, int CoutPad, int CinPad)
{
    int idx = blockIdx.x * blockDim.x + threadIdx.x;
    int total = KS * CoutPad * CinPad;
    if (idx >= total) return;
    int cin = idx % CinPad;
    int rem = idx / CinPad;
    int co = rem % CoutPad;
    int tap = rem / CoutPad;
    float v = (co < Cout && cin < Cin) ? W[((size_t)co * Cin + cin) * KS + tap] : 0.0f;
    W16[idx] = (_Float16)v;
}

// ---------------------------------------------------------------------------
// Fill reflect halo rows of a transposed-f16 feature buffer (B, T+2*PAD, Cpad).
// padded row p < PAD          copies from row 2*PAD - p
// padded row p >= T+PAD       copies from row 2T-2+2*PAD - p
// ---------------------------------------------------------------------------
__global__ void fill_halo(_Float16* __restrict__ yT, int B, int Cpad, int T)
{
    const int Tp = T + 2 * PAD;
    int idx = blockIdx.x * blockDim.x + threadIdx.x;
    int total = B * 2 * PAD * Cpad;
    if (idx >= total) return;
    int c = idx % Cpad;
    int rem = idx / Cpad;
    int pp = rem % (2 * PAD);
    int b = rem / (2 * PAD);
    int p   = (pp < PAD) ? pp : (T + pp);                    // dest padded row
    int src = (pp < PAD) ? (2 * PAD - pp) : (2 * T - 2 + 2 * PAD - p);
    yT[((size_t)b * Tp + p) * Cpad + c] = yT[((size_t)b * Tp + src) * Cpad + c];
}

// ---------------------------------------------------------------------------
// Conv1d + bias + leaky-relu as 7 tap-GEMMs on packed f16 data.
// Block = 128 threads = 4 wave32 tiles; wave w handles t-tile blockIdx.y*64+w*16.
// Inner loop per 16x16x32 MAC: 4x global_load_b128 + 1x v_wmma.
// Outputs: f32 (B,Cout,T) [guarded] and packed f16 transposed (B,Tp,CoutPad)
// ready to be the next layer's input (one v8h store per lane).
// ---------------------------------------------------------------------------
__global__ __launch_bounds__(128)
void conv_lrelu_wmma(const _Float16* __restrict__ xT,  // (B, Tp, CinPad)
                     const _Float16* __restrict__ W16, // (KS, CoutPad, CinPad)
                     const float* __restrict__ bias,
                     float* __restrict__ y32,          // (B, Cout, T)
                     _Float16* __restrict__ yT,        // (B, Tp, CoutPad) interior rows
                     int B, int CinPad, int Cout, int CoutPad, int T)
{
    const int Tp   = T + 2 * PAD;
    const int lane = threadIdx.x & 31;
    const int wv   = threadIdx.x >> 5;
    const int co0  = blockIdx.x * 16;
    const int t    = blockIdx.y * 64 + wv * 16 + (lane & 15);
    const int b    = blockIdx.z;
    // ISA 16-bit A layout: lanes 0-15 hold K {0..7,16..23}; lanes 16-31 +8
    const int baseA = (lane & 16) ? 8 : 0;
    // ISA 16-bit B layout: lanes 0-15 hold K {0..7,16..23}; lanes 16-31 +8
    const int baseB = (lane & 16) ? 8 : 0;

    const _Float16* wbase = W16 + (size_t)(co0 + (lane & 15)) * CinPad;
    const _Float16* xbase = xT + ((size_t)b * Tp + t) * CinPad;

    v8f acc = {};
#pragma unroll
    for (int tap = 0; tap < KS; ++tap) {
        const _Float16* wrow = wbase + (size_t)tap * CoutPad * CinPad;
        const _Float16* xrow = xbase + (size_t)tap * CinPad;   // padded row t+tap
        for (int c0 = 0; c0 < CinPad; c0 += 32) {
            v8h a0 = *(const v8h*)(wrow + c0 + baseA);
            v8h a1 = *(const v8h*)(wrow + c0 + baseA + 16);
            v8h b0 = *(const v8h*)(xrow + c0 + baseB);
            v8h b1 = *(const v8h*)(xrow + c0 + baseB + 16);
            acc = __builtin_amdgcn_wmma_f32_16x16x32_f16(
                      false, cat8(a0, a1), false, cat8(b0, b1),
                      (short)0, acc, false, false);
        }
    }

    // C layout: lane L -> N = L&15 (time t), vgpr r -> M = r + 8*(L>=16) (channel)
    const int half = lane >> 4;
    v8h pk;
#pragma unroll
    for (int r = 0; r < 8; ++r) {
        int co = co0 + r + 8 * half;
        float v = acc[r] + ((co < Cout) ? bias[co] : 0.0f);
        v = (v >= 0.0f) ? v : NEG_SLOPE * v;
        if (co < Cout) y32[((size_t)b * Cout + co) * T + t] = v;
        pk[r] = (_Float16)v;   // zero-pad channels store 0 (weights are zero there)
    }
    *(v8h*)(yT + ((size_t)b * Tp + (t + PAD)) * CoutPad + co0 + 8 * half) = pk;
}

// ---------------------------------------------------------------------------
// Per-position squared norms from the f32 features: nrm[b,t] = sum_c f^2
// ---------------------------------------------------------------------------
__global__ void norms_kernel(const float* __restrict__ feat, float* __restrict__ nrm,
                             int B, int C, int T)
{
    int idx = blockIdx.x * blockDim.x + threadIdx.x;
    if (idx >= B * T) return;
    int b = idx / T, t = idx - b * T;
    float s = 0.0f;
    for (int c = 0; c < C; ++c) {
        float v = feat[((size_t)b * C + c) * T + t];
        s += v * v;
    }
    nrm[idx] = s;
}

// ---------------------------------------------------------------------------
// dist[b,i,j] = sqrt(max(0, qn[i]+kn[j]-2*Gram)). Gram over H channels via
// WMMA on the packed transposed f16 features; global min/max via uint atomics
// (dist >= 0 so the float bit pattern is order-preserving).
// ---------------------------------------------------------------------------
__global__ __launch_bounds__(128)
void dist_wmma(const _Float16* __restrict__ qT, const _Float16* __restrict__ kT,
               const float* __restrict__ qn, const float* __restrict__ kn,
               float* __restrict__ dist, unsigned* __restrict__ minmax,
               int B, int Hp, int T)
{
    const int Tp   = T + 2 * PAD;
    const int lane = threadIdx.x & 31;
    const int wv   = threadIdx.x >> 5;
    const int i0 = blockIdx.x * 16;
    const int j0 = blockIdx.y * 64 + wv * 16;
    const int b  = blockIdx.z;
    const int baseA = (lane & 16) ? 8 : 0;
    const int baseB = (lane & 16) ? 8 : 0;

    const _Float16* qrow = qT + ((size_t)b * Tp + (i0 + (lane & 15) + PAD)) * Hp;
    const _Float16* krow = kT + ((size_t)b * Tp + (j0 + (lane & 15) + PAD)) * Hp;

    v8f acc = {};
    for (int c0 = 0; c0 < Hp; c0 += 32) {
        v8h a0 = *(const v8h*)(qrow + c0 + baseA);
        v8h a1 = *(const v8h*)(qrow + c0 + baseA + 16);
        v8h b0 = *(const v8h*)(krow + c0 + baseB);
        v8h b1 = *(const v8h*)(krow + c0 + baseB + 16);
        acc = __builtin_amdgcn_wmma_f32_16x16x32_f16(
                  false, cat8(a0, a1), false, cat8(b0, b1),
                  (short)0, acc, false, false);
    }

    unsigned lmin = 0x7F800000u, lmax = 0u;
    const int half = lane >> 4;
#pragma unroll
    for (int r = 0; r < 8; ++r) {
        int i = i0 + r + 8 * half;
        int j = j0 + (lane & 15);
        float d2 = qn[b * T + i] + kn[b * T + j] - 2.0f * acc[r];
        float d  = sqrtf(fmaxf(d2, 0.0f));
        dist[((size_t)b * T + i) * T + j] = d;
        unsigned u = __float_as_uint(d);
        lmin = (u < lmin) ? u : lmin;
        lmax = (u > lmax) ? u : lmax;
    }
    for (int off = 16; off > 0; off >>= 1) {
        unsigned om = (unsigned)__shfl_xor((int)lmin, off, 32);
        unsigned ox = (unsigned)__shfl_xor((int)lmax, off, 32);
        lmin = (om < lmin) ? om : lmin;
        lmax = (ox > lmax) ? ox : lmax;
    }
    if (lane == 0) {
        atomicMin(&minmax[0], lmin);
        atomicMax(&minmax[1], lmax);
    }
}

// ---------------------------------------------------------------------------
__global__ void init_kernel(float* __restrict__ path, unsigned* __restrict__ minmax, int n)
{
    int idx = blockIdx.x * blockDim.x + threadIdx.x;
    if (idx < n) path[idx] = 0.0f;
    if (idx == 0) { minmax[0] = 0x7F800000u; minmax[1] = 0u; }
}

__global__ void normalize_kernel(const float* __restrict__ draw,
                                 const unsigned* __restrict__ minmax,
                                 float* __restrict__ dout, int n)
{
    int idx = blockIdx.x * blockDim.x + threadIdx.x;
    if (idx >= n) return;
    float dmin = __uint_as_float(minmax[0]);
    float dmax = __uint_as_float(minmax[1]);
    float inv  = 1.0f / (dmax - dmin);
    dout[idx] = (draw[idx] - dmin) * inv;
}

// ---------------------------------------------------------------------------
// DTW anti-diagonal wavefront DP; one 512-thread workgroup per batch,
// rolling diagonals in LDS. move: 0=diag 1=left 2=up 3=stop
// ---------------------------------------------------------------------------
__global__ __launch_bounds__(512)
void dtw_moves_kernel(const float* __restrict__ dist, unsigned char* __restrict__ moves, int T)
{
    __shared__ float prev1[512];
    __shared__ float prev2[512];
    const int i = threadIdx.x;
    const int b = blockIdx.x;
    const float* D = dist + (size_t)b * T * T;
    unsigned char* Mv = moves + (size_t)b * T * T;

    prev1[i] = INF_F;
    prev2[i] = INF_F;
    __syncthreads();

    for (int d = 0; d < 2 * T - 1; ++d) {
        int j = d - i;
        float cur = INF_F;
        if (j >= 0 && j < T) {
            float dv   = D[(size_t)i * T + j];
            float diag = (i >= 1 && j >= 1) ? prev2[i - 1] : INF_F;
            float left = (j >= 1) ? prev1[i] : INF_F;
            float up   = (i >= 1) ? prev1[i - 1] : INF_F;
            float best = diag; int mv = 0;          // first-min tie-break
            if (left < best) { best = left; mv = 1; }
            if (up   < best) { best = up;   mv = 2; }
            if (i == 0 && j == 0) { cur = dv; mv = 3; }
            else                  { cur = best + dv; }
            Mv[(size_t)i * T + j] = (unsigned char)mv;
        }
        __syncthreads();
        prev2[i] = prev1[i];
        prev1[i] = cur;
        __syncthreads();
    }
}

__global__ void backtrack_kernel(const unsigned char* __restrict__ moves,
                                 const int* __restrict__ real_length,
                                 float* __restrict__ path, int T)
{
    if (threadIdx.x != 0) return;
    const int b = blockIdx.x;
    const unsigned char* Mv = moves + (size_t)b * T * T;
    float* P = path + (size_t)b * T * T;
    int rl = real_length[b];
    rl = (rl < 2) ? 2 : ((rl > T) ? T : rl);
    int ix = rl - 1, iy = rl - 1;
    for (int s = 0; s < 2 * T - 1; ++s) {
        P[(size_t)ix * T + iy] = 1.0f;
        int mv = Mv[(size_t)ix * T + iy];
        if (mv == 3) break;
        if (mv == 0)      { --ix; --iy; }
        else if (mv == 1) { --iy; }
        else              { --ix; }
    }
}

// ---------------------------------------------------------------------------
extern "C" void kernel_launch(void* const* d_in, const int* in_sizes, int n_in,
                              void* d_out, int out_size, void* d_ws, size_t ws_size,
                              hipStream_t stream)
{
    (void)in_sizes; (void)n_in; (void)out_size; (void)ws_size;
    const int B = 4, Cq = 63, Ck = 54, H = 128, T = 512;
    const int Tp = T + 2 * PAD;

    const float* vec   = (const float*)d_in[0];
    const float* music = (const float*)d_in[1];
    const int*   rlen  = (const int*)d_in[2];
    // d_in[3] = cal_path (1 in setup_inputs)
    const float* Wq1 = (const float*)d_in[4];  const float* bq1 = (const float*)d_in[5];
    const float* Wq2 = (const float*)d_in[6];  const float* bq2 = (const float*)d_in[7];
    const float* Wq3 = (const float*)d_in[8];  const float* bq3 = (const float*)d_in[9];
    const float* Wk1 = (const float*)d_in[10]; const float* bk1 = (const float*)d_in[11];
    const float* Wk2 = (const float*)d_in[12]; const float* bk2 = (const float*)d_in[13];
    const float* Wk3 = (const float*)d_in[14]; const float* bk3 = (const float*)d_in[15];

    // output: path01 (B*T*T) | dist (B*T*T) | q (B*H*T) | k (B*H*T)
    float* out      = (float*)d_out;
    float* path_out = out;
    float* dist_out = out + (size_t)B * T * T;
    float* q_out    = out + (size_t)2 * B * T * T;
    float* k_out    = q_out + (size_t)B * H * T;

    // ---- workspace (f32 section, 16B-aligned f16 section, byte section) ----
    float* f = (float*)d_ws;
    float* trash    = f;  f += (size_t)B * H * T;       // f32 sink for intermediate layers
    float* qn       = f;  f += (size_t)B * T;
    float* kn       = f;  f += (size_t)B * T;
    float* dist_raw = f;  f += (size_t)B * T * T;
    unsigned* minmax = (unsigned*)f; f += 4;            // 2 used + 2 pad (keeps 16B align)

    const size_t FBUF = (size_t)B * Tp * 128;           // halves, 16B multiple
    _Float16* h = (_Float16*)f;
    _Float16* xA  = h; h += FBUF;                       // packed input
    _Float16* xB  = h; h += FBUF;                       // layer-1 out
    _Float16* xC  = h; h += FBUF;                       // layer-2 out
    _Float16* qTf = h; h += FBUF;                       // final q features (f16,T-major)
    _Float16* kTf = h; h += FBUF;                       // final k features
    _Float16* wq1 = h; h += (size_t)KS * 64 * 64;
    _Float16* wq2 = h; h += (size_t)KS * 128 * 64;
    _Float16* wq3 = h; h += (size_t)KS * 128 * 128;
    _Float16* wk1 = h; h += (size_t)KS * 64 * 64;
    _Float16* wk2 = h; h += (size_t)KS * 128 * 64;
    _Float16* wk3 = h; h += (size_t)KS * 128 * 128;
    unsigned char* moves = (unsigned char*)h;           // B*T*T bytes

    auto cdiv = [](int a, int b) { return (a + b - 1) / b; };

    init_kernel<<<cdiv(B * T * T, 256), 256, 0, stream>>>(path_out, minmax, B * T * T);

    // pack weights (both paths)
    pack_weights<<<cdiv(KS * 64 * 64, 256), 256, 0, stream>>>(Wq1, wq1, Cq, Cq, 64, 64);
    pack_weights<<<cdiv(KS * 128 * 64, 256), 256, 0, stream>>>(Wq2, wq2, H, Cq, 128, 64);
    pack_weights<<<cdiv(KS * 128 * 128, 256), 256, 0, stream>>>(Wq3, wq3, H, H, 128, 128);
    pack_weights<<<cdiv(KS * 64 * 64, 256), 256, 0, stream>>>(Wk1, wk1, Ck, Ck, 64, 64);
    pack_weights<<<cdiv(KS * 128 * 64, 256), 256, 0, stream>>>(Wk2, wk2, H, Ck, 128, 64);
    pack_weights<<<cdiv(KS * 128 * 128, 256), 256, 0, stream>>>(Wk3, wk3, H, H, 128, 128);

    dim3 blk(128, 1, 1);
    // q encoder: 63 -> 63 -> 128 -> 128
    pack_input<<<cdiv(B * Tp * 64, 256), 256, 0, stream>>>(vec, xA, B, Cq, 64, T);
    conv_lrelu_wmma<<<dim3(64 / 16, T / 64, B), blk, 0, stream>>>(xA, wq1, bq1, trash, xB, B, 64, Cq, 64, T);
    fill_halo<<<cdiv(B * 2 * PAD * 64, 256), 256, 0, stream>>>(xB, B, 64, T);
    conv_lrelu_wmma<<<dim3(128 / 16, T / 64, B), blk, 0, stream>>>(xB, wq2, bq2, trash, xC, B, 64, H, 128, T);
    fill_halo<<<cdiv(B * 2 * PAD * 128, 256), 256, 0, stream>>>(xC, B, 128, T);
    conv_lrelu_wmma<<<dim3(128 / 16, T / 64, B), blk, 0, stream>>>(xC, wq3, bq3, q_out, qTf, B, 128, H, 128, T);

    // k encoder: 54 -> 54 -> 128 -> 128
    pack_input<<<cdiv(B * Tp * 64, 256), 256, 0, stream>>>(music, xA, B, Ck, 64, T);
    conv_lrelu_wmma<<<dim3(64 / 16, T / 64, B), blk, 0, stream>>>(xA, wk1, bk1, trash, xB, B, 64, Ck, 64, T);
    fill_halo<<<cdiv(B * 2 * PAD * 64, 256), 256, 0, stream>>>(xB, B, 64, T);
    conv_lrelu_wmma<<<dim3(128 / 16, T / 64, B), blk, 0, stream>>>(xB, wk2, bk2, trash, xC, B, 64, H, 128, T);
    fill_halo<<<cdiv(B * 2 * PAD * 128, 256), 256, 0, stream>>>(xC, B, 128, T);
    conv_lrelu_wmma<<<dim3(128 / 16, T / 64, B), blk, 0, stream>>>(xC, wk3, bk3, k_out, kTf, B, 128, H, 128, T);

    // norms from the f32 outputs
    norms_kernel<<<cdiv(B * T, 256), 256, 0, stream>>>(q_out, qn, B, H, T);
    norms_kernel<<<cdiv(B * T, 256), 256, 0, stream>>>(k_out, kn, B, H, T);

    // distance matrix + global min/max
    dist_wmma<<<dim3(T / 16, T / 64, B), blk, 0, stream>>>(qTf, kTf, qn, kn, dist_raw, minmax, B, 128, T);

    normalize_kernel<<<cdiv(B * T * T, 256), 256, 0, stream>>>(dist_raw, minmax, dist_out, B * T * T);

    dtw_moves_kernel<<<B, 512, 0, stream>>>(dist_out, moves, T);
    backtrack_kernel<<<B, 32, 0, stream>>>(moves, rlen, path_out, T);
}
